// LayerNormMoELayer_15032385536475
// MI455X (gfx1250) — compile-verified, matrix operand order checked by reading, not run on hardware
//
#include <hip/hip_runtime.h>
#include <hip/hip_bf16.h>

// ---------------------------------------------------------------------------
// MoE layer (RMSNorm -> top-2 router -> gathered SwiGLU experts) for gfx1250.
// bf16 WMMA (v_wmma_f32_16x16x32_bf16) for all three GEMMs; fp32 weights
// converted to bf16 while staging into LDS (weight read = HBM roofline floor).
// K-tile = 64 (2 WMMA K-steps per LDS stage); bf16 activation tiles staged via
// GLOBAL_LOAD_ASYNC_TO_LDS (ASYNCcnt) when the builtin exists.
// ---------------------------------------------------------------------------

#define T_TOK 4096
#define HDIM  1024
#define FDIM  2816
#define NEXP  8
#define RMS_EPS 1e-5f

typedef __bf16 bf16;
typedef __attribute__((ext_vector_type(16))) __bf16 v16bf;
typedef __attribute__((ext_vector_type(8)))  float  v8f;
typedef __attribute__((ext_vector_type(4)))  __bf16 v4bf;
typedef int v4i __attribute__((vector_size(16)));   // matches builtin prototype

// LDS tile row stride (elements) for 64-wide K tiles; padded vs bank conflicts
#define KS 72

#if defined(__gfx1250__) && __has_builtin(__builtin_amdgcn_global_load_async_to_lds_b128)
#define USE_ASYNC_LDS 1
#else
#define USE_ASYNC_LDS 0
#endif

#if USE_ASYNC_LDS
#define ASYNC_CP16(dst, src)                                              \
  __builtin_amdgcn_global_load_async_to_lds_b128(                         \
      (__attribute__((address_space(1))) v4i*)(src),                      \
      (__attribute__((address_space(3))) v4i*)(dst), 0, 0)
__device__ inline void async_wait0() {
#if __has_builtin(__builtin_amdgcn_s_wait_asynccnt)
  __builtin_amdgcn_s_wait_asynccnt(0);
#else
  asm volatile("s_wait_asynccnt 0" ::: "memory");
#endif
}
#endif

__device__ inline float fast_sigmoid(float g) {
  float d = 1.0f + __expf(-g);
#if __has_builtin(__builtin_amdgcn_rcpf)
  return __builtin_amdgcn_rcpf(d);          // single v_rcp_f32, no div-scale chain
#else
  return 1.0f / d;
#endif
}

// ---------------------------------------------------------------------------
// Kernel 1: RMSNorm + fp32->bf16. One 256-thread block per token.
// ---------------------------------------------------------------------------
__global__ __launch_bounds__(256)
void rmsnorm_kernel(const float* __restrict__ x, const float* __restrict__ lnw,
                    bf16* __restrict__ xb) {
  const int t = blockIdx.x, tid = threadIdx.x;
  const float* row = x + (size_t)t * HDIM;
  float4 v = ((const float4*)row)[tid];            // 4 elems / thread
  float ss = v.x * v.x + v.y * v.y + v.z * v.z + v.w * v.w;
#pragma unroll
  for (int off = 16; off > 0; off >>= 1) ss += __shfl_xor(ss, off, 32);

  __shared__ float red[8];
  const int wid = tid >> 5, lane = tid & 31;
  if (lane == 0) red[wid] = ss;
  __syncthreads();
  if (tid == 0) {
    float tot = 0.f;
#pragma unroll
    for (int i = 0; i < 8; ++i) tot += red[i];
    red[0] = rsqrtf(tot / (float)HDIM + RMS_EPS);
  }
  __syncthreads();
  const float scale = red[0];
  float4 w = ((const float4*)lnw)[tid];
  v4bf o;
  o[0] = (bf16)(v.x * scale * w.x);
  o[1] = (bf16)(v.y * scale * w.y);
  o[2] = (bf16)(v.z * scale * w.z);
  o[3] = (bf16)(v.w * scale * w.w);
  ((v4bf*)xb)[(size_t)t * (HDIM / 4) + tid] = o;
}

// ---------------------------------------------------------------------------
// Kernel 2: router. One wave per token: logits = x @ Wr, softmax, top-2,
// append (token, prob) to each selected expert's list.
// ---------------------------------------------------------------------------
__global__ __launch_bounds__(256)
void router_kernel(const bf16* __restrict__ xb, const float* __restrict__ rw,
                   int* __restrict__ counts, int* __restrict__ tok,
                   float* __restrict__ tokw) {
  const int wid = threadIdx.x >> 5, lane = threadIdx.x & 31;
  const int t = blockIdx.x * 8 + wid;
  float acc[NEXP];
#pragma unroll
  for (int e = 0; e < NEXP; ++e) acc[e] = 0.f;

  for (int h = lane; h < HDIM; h += 32) {
    float xv = (float)xb[(size_t)t * HDIM + h];
    const float4 r0 = ((const float4*)(rw + (size_t)h * NEXP))[0];
    const float4 r1 = ((const float4*)(rw + (size_t)h * NEXP))[1];
    acc[0] += xv * r0.x; acc[1] += xv * r0.y; acc[2] += xv * r0.z; acc[3] += xv * r0.w;
    acc[4] += xv * r1.x; acc[5] += xv * r1.y; acc[6] += xv * r1.z; acc[7] += xv * r1.w;
  }
#pragma unroll
  for (int off = 16; off > 0; off >>= 1)
#pragma unroll
    for (int e = 0; e < NEXP; ++e) acc[e] += __shfl_xor(acc[e], off, 32);

  if (lane == 0) {
    float mx = acc[0];
#pragma unroll
    for (int e = 1; e < NEXP; ++e) mx = fmaxf(mx, acc[e]);
    float p[NEXP], s = 0.f;
#pragma unroll
    for (int e = 0; e < NEXP; ++e) { p[e] = __expf(acc[e] - mx); s += p[e]; }
    const float inv = 1.f / s;
#pragma unroll
    for (int e = 0; e < NEXP; ++e) p[e] *= inv;
    int i1 = 0;
#pragma unroll
    for (int e = 1; e < NEXP; ++e) if (p[e] > p[i1]) i1 = e;
    int i2 = (i1 == 0) ? 1 : 0;
#pragma unroll
    for (int e = 0; e < NEXP; ++e) if (e != i1 && p[e] > p[i2]) i2 = e;
    int s1 = atomicAdd(&counts[i1], 1);
    tok[i1 * T_TOK + s1] = t; tokw[i1 * T_TOK + s1] = p[i1];
    int s2 = atomicAdd(&counts[i2], 1);
    tok[i2 * T_TOK + s2] = t; tokw[i2 * T_TOK + s2] = p[i2];
  }
}

// load a 16x32 bf16 A/B fragment from an LDS tile with row stride KS
__device__ inline v16bf load_frag(const bf16* __restrict__ base, int row, int kb) {
  v16bf f;
  const bf16* p = base + row * KS + kb;
#pragma unroll
  for (int j = 0; j < 8; ++j) { f[j] = p[j]; f[8 + j] = p[16 + j]; }
  return f;
}

// ---------------------------------------------------------------------------
// Kernel 3 (per expert): gathered gate/up GEMM + SiLU*up*routeW -> Hh (bf16).
// Block tile: 128 gathered rows x 64 F-cols for BOTH gate and up (X reuse).
// 8 waves in a 4x2 grid; each wave owns 32x32 per matrix (2x2 WMMA tiles).
// K-tile 64 => 16 WMMA per wave per barrier pair.
// ---------------------------------------------------------------------------
__global__ __launch_bounds__(256)
void moe_gateup_kernel(const bf16* __restrict__ xb, const float* __restrict__ wg,
                       const float* __restrict__ wu, const int* __restrict__ tok,
                       const float* __restrict__ tokw, const int* __restrict__ counts,
                       bf16* __restrict__ Hh, int e) {
  const int count = counts[e];
  const int m0 = blockIdx.y * 128;
  if (m0 >= count) return;                       // dynamic expert load: early-exit
  const int n0 = blockIdx.x * 64;
  const int tid = threadIdx.x;

  __shared__ __align__(16) bf16 sX[128 * KS];
  __shared__ __align__(16) bf16 sG[64 * KS];
  __shared__ __align__(16) bf16 sU[64 * KS];
  __shared__ float sW[128];
  __shared__ int   sT[128];

  if (tid < 128) {
    const int gm = m0 + tid;
    int t = 0; float w = 0.f;
    if (gm < count) { t = tok[e * T_TOK + gm]; w = tokw[e * T_TOK + gm]; }
    sT[tid] = t; sW[tid] = w;                    // padded rows get weight 0
  }
  __syncthreads();

  const int wid = tid >> 5, lane = tid & 31;
  const int wm = wid >> 1, wn = wid & 1;         // 4x2 wave grid
  const int lrow = lane & 15, lhalf = lane >> 4;

  v8f accG[2][2] = {}; v8f accU[2][2] = {};

  const int xrow = tid >> 1, xseg = tid & 1;     // X staging: 32 bf16 / thread
  const int wk2 = (tid >> 3) * 2;                // W staging: 2 k-rows x 8 cols
  const int wnc = (tid & 7) * 8;
  const bf16* xsrc_base = xb + (size_t)sT[xrow] * HDIM + xseg * 32;

  for (int k0 = 0; k0 < HDIM; k0 += 64) {
    // stage X tile [128 x 64] (bf16, gathered rows) -- async when available
    {
      const bf16* src = xsrc_base + k0;
      bf16* dst = &sX[xrow * KS + xseg * 32];
#if USE_ASYNC_LDS
#pragma unroll
      for (int j = 0; j < 4; ++j) ASYNC_CP16(dst + j * 8, src + j * 8);
#else
#pragma unroll
      for (int j = 0; j < 4; ++j) ((uint4*)dst)[j] = ((const uint4*)src)[j];
#endif
    }
    // stage Wg/Wu tiles [64 x 64], fp32 -> bf16, TRANSPOSED into LDS [64][64]
#pragma unroll
    for (int r = 0; r < 2; ++r) {
      const size_t krow = (size_t)e * HDIM + (k0 + wk2 + r);
      const float* gsrc = wg + krow * FDIM + n0 + wnc;
      const float* usrc = wu + krow * FDIM + n0 + wnc;
      float4 g0 = ((const float4*)gsrc)[0], g1 = ((const float4*)gsrc)[1];
      float4 u0 = ((const float4*)usrc)[0], u1 = ((const float4*)usrc)[1];
#pragma unroll
      for (int j = 0; j < 4; ++j) {
        sG[(wnc + j)     * KS + wk2 + r] = (bf16)(&g0.x)[j];
        sG[(wnc + 4 + j) * KS + wk2 + r] = (bf16)(&g1.x)[j];
        sU[(wnc + j)     * KS + wk2 + r] = (bf16)(&u0.x)[j];
        sU[(wnc + 4 + j) * KS + wk2 + r] = (bf16)(&u1.x)[j];
      }
      __builtin_prefetch(gsrc + 64 * FDIM, 0, 1);  // next k-tile
      __builtin_prefetch(usrc + 64 * FDIM, 0, 1);
    }
#if USE_ASYNC_LDS
    async_wait0();
#endif
    __syncthreads();

#pragma unroll
    for (int ks = 0; ks < 64; ks += 32) {
      const int kb = ks + lhalf * 8;
      v16bf a[2], bg[2], bu[2];
#pragma unroll
      for (int mi = 0; mi < 2; ++mi)
        a[mi] = load_frag(sX, wm * 32 + mi * 16 + lrow, kb);
#pragma unroll
      for (int ni = 0; ni < 2; ++ni) {
        bg[ni] = load_frag(sG, wn * 32 + ni * 16 + lrow, kb);
        bu[ni] = load_frag(sU, wn * 32 + ni * 16 + lrow, kb);
      }
#pragma unroll
      for (int mi = 0; mi < 2; ++mi)
#pragma unroll
        for (int ni = 0; ni < 2; ++ni) {
          accG[mi][ni] = __builtin_amdgcn_wmma_f32_16x16x32_bf16(
              false, a[mi], false, bg[ni], (short)0, accG[mi][ni], false, false);
          accU[mi][ni] = __builtin_amdgcn_wmma_f32_16x16x32_bf16(
              false, a[mi], false, bu[ni], (short)0, accU[mi][ni], false, false);
        }
    }
    __syncthreads();
  }

  // epilogue: h = silu(gate) * up * route_weight, store bf16 (gathered rows)
#pragma unroll
  for (int mi = 0; mi < 2; ++mi)
#pragma unroll
    for (int ni = 0; ni < 2; ++ni)
#pragma unroll
      for (int i = 0; i < 8; ++i) {
        const int m = wm * 32 + mi * 16 + i + 8 * lhalf;   // C layout: M=i(+8)
        const int n = n0 + wn * 32 + ni * 16 + lrow;       // N=lane&15
        float g = accG[mi][ni][i], u = accU[mi][ni][i];
        float h = g * fast_sigmoid(g) * u * sW[m];
        Hh[(size_t)(m0 + m) * FDIM + n] = (bf16)h;
      }
}

// ---------------------------------------------------------------------------
// Kernel 4 (per expert): down projection. out[token] += Hh @ Wd  (route weight
// already folded into Hh). Sequential per-expert launches -> plain fp32 RMW,
// no atomics, deterministic.
// ---------------------------------------------------------------------------
__global__ __launch_bounds__(256)
void moe_down_kernel(const bf16* __restrict__ Hh, const float* __restrict__ wd,
                     const int* __restrict__ tok, const int* __restrict__ counts,
                     float* __restrict__ out, int e) {
  const int count = counts[e];
  const int m0 = blockIdx.y * 128;
  if (m0 >= count) return;
  const int n0 = blockIdx.x * 64;
  const int tid = threadIdx.x;

  __shared__ __align__(16) bf16 sA[128 * KS];
  __shared__ __align__(16) bf16 sB[64 * KS];
  __shared__ int sT[128];

  if (tid < 128) {
    const int gm = m0 + tid;
    sT[tid] = (gm < count) ? tok[e * T_TOK + gm] : 0;
  }
  __syncthreads();

  const int wid = tid >> 5, lane = tid & 31;
  const int wm = wid >> 1, wn = wid & 1;
  const int lrow = lane & 15, lhalf = lane >> 4;

  v8f acc[2][2] = {};

  const int arow = tid >> 1, aseg = tid & 1;
  const int wk2 = (tid >> 3) * 2, wnc = (tid & 7) * 8;

  for (int k0 = 0; k0 < FDIM; k0 += 64) {
    // stage Hh tile [128 x 64] (bf16, gathered rows already) -- async path
    {
      const bf16* src = Hh + (size_t)(m0 + arow) * FDIM + k0 + aseg * 32;
      bf16* dst = &sA[arow * KS + aseg * 32];
#if USE_ASYNC_LDS
#pragma unroll
      for (int j = 0; j < 4; ++j) ASYNC_CP16(dst + j * 8, src + j * 8);
#else
#pragma unroll
      for (int j = 0; j < 4; ++j) ((uint4*)dst)[j] = ((const uint4*)src)[j];
#endif
    }
    // stage Wd tile [64 x 64], fp32 -> bf16, transposed
#pragma unroll
    for (int r = 0; r < 2; ++r) {
      const float* bsrc = wd + ((size_t)e * FDIM + (k0 + wk2 + r)) * HDIM + n0 + wnc;
      float4 b0 = ((const float4*)bsrc)[0], b1 = ((const float4*)bsrc)[1];
#pragma unroll
      for (int j = 0; j < 4; ++j) {
        sB[(wnc + j)     * KS + wk2 + r] = (bf16)(&b0.x)[j];
        sB[(wnc + 4 + j) * KS + wk2 + r] = (bf16)(&b1.x)[j];
      }
      __builtin_prefetch(bsrc + 64 * HDIM, 0, 1);
    }
#if USE_ASYNC_LDS
    async_wait0();
#endif
    __syncthreads();

#pragma unroll
    for (int ks = 0; ks < 64; ks += 32) {
      const int kb = ks + lhalf * 8;
      v16bf a[2], b[2];
#pragma unroll
      for (int mi = 0; mi < 2; ++mi)
        a[mi] = load_frag(sA, wm * 32 + mi * 16 + lrow, kb);
#pragma unroll
      for (int ni = 0; ni < 2; ++ni)
        b[ni] = load_frag(sB, wn * 32 + ni * 16 + lrow, kb);
#pragma unroll
      for (int mi = 0; mi < 2; ++mi)
#pragma unroll
        for (int ni = 0; ni < 2; ++ni)
          acc[mi][ni] = __builtin_amdgcn_wmma_f32_16x16x32_bf16(
              false, a[mi], false, b[ni], (short)0, acc[mi][ni], false, false);
    }
    __syncthreads();
  }

  // scatter-accumulate into out (this block owns these (token, n) cells)
#pragma unroll
  for (int mi = 0; mi < 2; ++mi)
#pragma unroll
    for (int ni = 0; ni < 2; ++ni)
#pragma unroll
      for (int i = 0; i < 8; ++i) {
        const int m = wm * 32 + mi * 16 + i + 8 * lhalf;
        if (m0 + m < count) {
          const int t = sT[m];
          const int n = n0 + wn * 32 + ni * 16 + lrow;
          float* p = out + (size_t)t * HDIM + n;
          *p = *p + acc[mi][ni][i];
        }
      }
}

// ---------------------------------------------------------------------------
extern "C" void kernel_launch(void* const* d_in, const int* in_sizes, int n_in,
                              void* d_out, int out_size, void* d_ws, size_t ws_size,
                              hipStream_t stream) {
  (void)in_sizes; (void)n_in; (void)out_size; (void)ws_size;
  const float* x   = (const float*)d_in[0];
  const float* lnw = (const float*)d_in[1];
  const float* rw  = (const float*)d_in[2];
  const float* wg  = (const float*)d_in[3];
  const float* wu  = (const float*)d_in[4];
  const float* wd  = (const float*)d_in[5];
  float* out = (float*)d_out;

  char* ws = (char*)d_ws;
  bf16*  xb     = (bf16*)ws;                                          // 8 MB
  int*   counts = (int*)(ws + (size_t)8388608);                       // 256 B
  int*   tok    = (int*)(ws + (size_t)8388864);                       // 128 KB
  float* tokw   = (float*)(ws + (size_t)8519936);                     // 128 KB
  bf16*  Hh     = (bf16*)(ws + (size_t)8651008);                      // 23 MB (reused per expert)

  (void)hipMemsetAsync(counts, 0, 256, stream);
  (void)hipMemsetAsync(out, 0, (size_t)T_TOK * HDIM * sizeof(float), stream);

  rmsnorm_kernel<<<T_TOK, 256, 0, stream>>>(x, lnw, xb);
  router_kernel<<<T_TOK / 8, 256, 0, stream>>>(xb, rw, counts, tok, tokw);

  for (int e = 0; e < NEXP; ++e) {
    moe_gateup_kernel<<<dim3(FDIM / 64, T_TOK / 128), 256, 0, stream>>>(
        xb, wg, wu, tok, tokw, counts, Hh, e);
    moe_down_kernel<<<dim3(HDIM / 64, T_TOK / 128), 256, 0, stream>>>(
        Hh, wd, tok, counts, out, e);
  }
}
